// SEBlock_9612136808477
// MI455X (gfx1250) — matrix operand order repaired and checked by reading, not compiled, not run on hardware
//
#include <hip/hip_runtime.h>
#include <hip/hip_bf16.h>

typedef __attribute__((ext_vector_type(16))) _Float16 v16h;
typedef __attribute__((ext_vector_type(8)))  _Float16 v8h;
typedef __attribute__((ext_vector_type(8)))  float    v8f;

#define NPTS   100000
#define CCH    128
#define KOFF   27
#define MPTS   60000
#define NBATCH 8
#define CRED   32
#define BN_EPS 1e-5f
#define MTILES (MPTS / 16)        // 3750 16-row tiles per kernel offset
#define TPB    6                  // m-tiles per block (3750 = 6 * 625)
#define BLKS_PER_K (MTILES / TPB) // 625

__device__ __forceinline__ float gAtomAdd(float* p, float v) {
  return __hip_atomic_fetch_add(p, v, __ATOMIC_RELAXED, __HIP_MEMORY_SCOPE_AGENT);
}
__device__ __forceinline__ float sAtomAdd(float* p, float v) {
  return __hip_atomic_fetch_add(p, v, __ATOMIC_RELAXED, __HIP_MEMORY_SCOPE_WORKGROUP);
}

// ---------------------------------------------------------------- utilities
__global__ void __launch_bounds__(256)
zero_f32(float* __restrict__ p, int n) {
  for (int i = blockIdx.x * 256 + threadIdx.x; i < n; i += gridDim.x * 256)
    p[i] = 0.f;
}

// W[k,c,d] (fp32, d contiguous)  ->  Wcol[k, d*128 + c] (f16, c contiguous)
// i.e. column-major per output channel: matches the contiguous-K B fragments conv_wmma loads.
__global__ void __launch_bounds__(256)
convert_w(const float* __restrict__ W, _Float16* __restrict__ Wcol) {
  const int total = KOFF * CCH * CCH;
  for (int idx = blockIdx.x * 256 + threadIdx.x; idx < total; idx += gridDim.x * 256) {
    int k   = idx >> 14;          // /16384
    int rem = idx & 16383;
    int c   = rem >> 7;           // input channel
    int d   = rem & 127;          // output channel
    Wcol[(k << 14) + (d << 7) + c] = (_Float16)W[idx];
  }
}

// ------------------------------------------------- gather-WMMA-scatter conv
// One block = one kernel offset k and TPB consecutive 16-row m-tiles.
// B-fragments (the 128x128 weight slab for this k, this wave's 16-col strip)
// are hoisted into registers once and reused across all TPB tiles.
// A is double-buffered in LDS: gathers for tile t+1 are issued before the
// WMMAs of tile t so their latency hides behind the matrix ops.
__global__ void __launch_bounds__(256)
conv_wmma(const float* __restrict__ xin,
          const _Float16* __restrict__ wcol,
          const int* __restrict__ in_idx,
          const int* __restrict__ out_idx,
          const float* __restrict__ se,        // nullable: SE scale [B, C]
          const int* __restrict__ batch_id,    // nullable (with se)
          float* __restrict__ acc) {
  __shared__ _Float16 As[2][16][136];          // ping-pong; 68 DWORDs/row, bank-tiled
  __shared__ int rin[TPB * 16], rout[TPB * 16];

  const int k    = blockIdx.x / BLKS_PER_K;
  const int m00  = (blockIdx.x % BLKS_PER_K) * (TPB * 16);
  const int tid  = threadIdx.x;
  const int base = k * MPTS + m00;

  // Stage all TPB*16 gather/scatter indices once.
  if (tid < TPB * 16)                       rin[tid]        = in_idx [base + tid];
  else if (tid >= 128 && tid < 128 + TPB * 16) rout[tid - 128] = out_idx[base + tid - 128];

  const int wave = tid >> 5;
  const int lane = tid & 31;
  const int hi   = lane >> 4;                  // lane half selects K sub-span
  const int rA   = lane & 15;                  // A row
  const int n0   = wave * 16 + (lane & 15);    // output column
  const _Float16* wk = wcol + ((size_t)k << 14) + ((size_t)n0 << 7);

  // Hoisted B fragments: 4 K-steps x 16 halves = 32 VGPRs, reused TPB times.
  v16h barr[4];
#pragma unroll
  for (int kk = 0; kk < 4; kk++) {
    const int kbB = kk * 32 + hi * 16;         // 16 contiguous K for column n0
    v8h b0 = *(const v8h*)(wk + kbB);
    v8h b1 = *(const v8h*)(wk + kbB + 8);
    barr[kk] = __builtin_shufflevector(b0, b1, 0,1,2,3,4,5,6,7,8,9,10,11,12,13,14,15);
  }

  const int gr = tid >> 4;                     // gather row 0..15
  const int gc = (tid & 15) << 3;              // gather col base (8 floats)

  __syncthreads();                             // indices visible

  // Prologue: gather tile 0 into As[0].
  {
    const int row = rin[gr];
    const float* src = xin + (size_t)row * CCH + gc;
    float v[8];
    if (se) {
      const float* sp = se + (size_t)batch_id[row] * CCH + gc;
#pragma unroll
      for (int i = 0; i < 8; i++) v[i] = src[i] * sp[i];
    } else {
#pragma unroll
      for (int i = 0; i < 8; i++) v[i] = src[i];
    }
#pragma unroll
    for (int i = 0; i < 8; i++) As[0][gr][gc + i] = (_Float16)v[i];
  }

  for (int t = 0; t < TPB; ++t) {
    __syncthreads();                           // As[t&1] fully written

    // Issue gathers for tile t+1 now; consume (convert+LDS store) after compute.
    const bool more = (t + 1 < TPB);           // uniform branch: EXEC stays full
    float nxt[8];
    if (more) {
      const int row = rin[(t + 1) * 16 + gr];
      const float* src = xin + (size_t)row * CCH + gc;
      if (se) {
        const float* sp = se + (size_t)batch_id[row] * CCH + gc;
#pragma unroll
        for (int i = 0; i < 8; i++) nxt[i] = src[i] * sp[i];
      } else {
#pragma unroll
        for (int i = 0; i < 8; i++) nxt[i] = src[i];
      }
    }

    // 16x128 output tile: this wave's 16-col strip, 4 WMMAs over C=128.
    const _Float16 (*Ab)[136] = As[t & 1];
    v8f cacc = {};
#pragma unroll
    for (int kk = 0; kk < 4; kk++) {
      const int kbA = kk * 32 + hi * 16;       // A: K = kbA..+15 contiguous
      v8h a0 = *(const v8h*)&Ab[rA][kbA];
      v8h a1 = *(const v8h*)&Ab[rA][kbA + 8];
      v16h a = __builtin_shufflevector(a0, a1, 0,1,2,3,4,5,6,7,8,9,10,11,12,13,14,15);
      cacc = __builtin_amdgcn_wmma_f32_16x16x32_f16(
          /*neg_a=*/false, a, /*neg_b=*/false, barr[kk],
          /*c_mod=*/(short)0, cacc, /*reuse_a=*/false, /*reuse_b=*/false);
    }

    // D layout: VGPR v -> row v + 8*hi, col lane&15. Scatter-add.
    const int mb = t * 16 + hi * 8;
#pragma unroll
    for (int v = 0; v < 8; v++) {
      const int ridx = rout[mb + v];
      gAtomAdd(&acc[(size_t)ridx * CCH + n0], cacc[v]);
    }

    // Convert + store the prefetched tile into the other LDS buffer.
    if (more) {
      _Float16 (*An)[136] = As[(t + 1) & 1];
#pragma unroll
      for (int i = 0; i < 8; i++) An[gr][gc + i] = (_Float16)nxt[i];
    }
  }
}

// --------------------------------------------------------------- batch norm
__global__ void __launch_bounds__(256)
bn_stats(const float* __restrict__ acc, float* __restrict__ sums,
         float* __restrict__ sumsq) {
  __shared__ float sh1[256], sh2[256];
  const int tid = threadIdx.x;
  const int c = tid & 127, rsub = tid >> 7;
  const int per = (NPTS + gridDim.x - 1) / gridDim.x;
  const int r0 = blockIdx.x * per;
  const int r1 = (r0 + per < NPTS) ? (r0 + per) : NPTS;
  float s = 0.f, q = 0.f;
  for (int r = r0 + rsub; r < r1; r += 2) {
    float v = acc[(size_t)r * CCH + c];
    s += v; q += v * v;
  }
  sh1[tid] = s; sh2[tid] = q;
  __syncthreads();
  if (tid < 128) {
    gAtomAdd(&sums[c],  sh1[tid] + sh1[tid + 128]);
    gAtomAdd(&sumsq[c], sh2[tid] + sh2[tid + 128]);
  }
}

// BN1 + ReLU, write activations, and LDS-reduced per-batch segment pooling.
__global__ void __launch_bounds__(256)
bn_apply_pool(const float* __restrict__ acc, const float* __restrict__ sums,
              const float* __restrict__ sumsq, const float* __restrict__ gamma,
              const float* __restrict__ beta, const int* __restrict__ batch_id,
              float* __restrict__ act, float* __restrict__ pooled) {
  __shared__ float lpool[NBATCH * CCH];
  const int tid = threadIdx.x;
  for (int i = tid; i < NBATCH * CCH; i += 256) lpool[i] = 0.f;
  __syncthreads();

  const int c = tid & 127, rsub = tid >> 7;
  const float mean = sums[c] * (1.f / NPTS);
  const float var  = sumsq[c] * (1.f / NPTS) - mean * mean;
  const float inv  = rsqrtf(var + BN_EPS);
  const float g = gamma[c], b = beta[c];

  const int per = (NPTS + gridDim.x - 1) / gridDim.x;
  const int r0 = blockIdx.x * per;
  const int r1 = (r0 + per < NPTS) ? (r0 + per) : NPTS;
  for (int r = r0 + rsub; r < r1; r += 2) {
    float v = acc[(size_t)r * CCH + c];
    float y = fmaxf((v - mean) * inv * g + b, 0.f);
    act[(size_t)r * CCH + c] = y;
    sAtomAdd(&lpool[batch_id[r] * CCH + c], y);
  }
  __syncthreads();
  for (int i = tid; i < NBATCH * CCH; i += 256) gAtomAdd(&pooled[i], lpool[i]);
}

// ------------------------------------------------------------------- SE MLP
__global__ void __launch_bounds__(256)
se_mlp(const float* __restrict__ pooled, const float* __restrict__ fc1,
       const float* __restrict__ fc2, float* __restrict__ seg) {
  __shared__ float h[NBATCH * CRED];
  const int tid = threadIdx.x;
  {  // fc1 + relu : 256 threads = 8 batches x 32 hidden
    const int bb = tid >> 5, j = tid & 31;
    float s = 0.f;
    for (int cc = 0; cc < CCH; cc++) s += pooled[bb * CCH + cc] * fc1[cc * CRED + j];
    h[bb * CRED + j] = fmaxf(s, 0.f);
  }
  __syncthreads();
  for (int idx = tid; idx < NBATCH * CCH; idx += 256) {  // fc2 + sigmoid
    const int bb = idx >> 7, d = idx & 127;
    float s = 0.f;
    for (int j = 0; j < CRED; j++) s += h[bb * CRED + j] * fc2[j * CCH + d];
    seg[idx] = 1.f / (1.f + __expf(-s));
  }
}

// ----------------------------------------------- BN2 + residual + ReLU out
__global__ void __launch_bounds__(256)
bn_final(const float* __restrict__ acc, const float* __restrict__ xin,
         const float* __restrict__ sums, const float* __restrict__ sumsq,
         const float* __restrict__ gamma, const float* __restrict__ beta,
         float* __restrict__ out) {
  const size_t total = (size_t)NPTS * CCH;
  const size_t stride = (size_t)gridDim.x * 256;
  for (size_t i = blockIdx.x * 256 + threadIdx.x; i < total; i += stride) {
    const int c = (int)(i & 127);
    const float mean = sums[c] * (1.f / NPTS);
    const float inv  = rsqrtf(sumsq[c] * (1.f / NPTS) - mean * mean + BN_EPS);
    float y = (acc[i] - mean) * inv * gamma[c] + beta[c] + xin[i];
    out[i] = fmaxf(y, 0.f);
  }
}

// ------------------------------------------------------------------ launch
extern "C" void kernel_launch(void* const* d_in, const int* in_sizes, int n_in,
                              void* d_out, int out_size, void* d_ws, size_t ws_size,
                              hipStream_t stream) {
  const float* x      = (const float*)d_in[0];
  const int*   bidv   = (const int*)  d_in[1];
  const int*   in_idx = (const int*)  d_in[2];
  const int*   outix  = (const int*)  d_in[3];
  const float* W1     = (const float*)d_in[4];
  const float* W2     = (const float*)d_in[5];
  const float* fc1_w  = (const float*)d_in[6];
  const float* fc2_w  = (const float*)d_in[7];
  const float* gamma1 = (const float*)d_in[8];
  const float* beta1  = (const float*)d_in[9];
  const float* gamma2 = (const float*)d_in[10];
  const float* beta2  = (const float*)d_in[11];
  float* out = (float*)d_out;

  char* ws = (char*)d_ws;
  size_t off = 0;
  _Float16* Wcol1 = (_Float16*)(ws + off); off += (size_t)KOFF * CCH * CCH * 2;   // 884736 B
  _Float16* Wcol2 = (_Float16*)(ws + off); off += (size_t)KOFF * CCH * CCH * 2;
  float* acc = (float*)(ws + off);         off += (size_t)NPTS * CCH * 4;         // 51.2 MB
  float* act = (float*)(ws + off);         off += (size_t)NPTS * CCH * 4;         // 51.2 MB
  float* stats = (float*)(ws + off);       off += 2560 * 4;
  if (ws_size < off) return;  // workspace too small: refuse to scribble

  float* sums1  = stats;
  float* sumsq1 = stats + 128;
  float* sums2  = stats + 256;
  float* sumsq2 = stats + 384;
  float* pooled = stats + 512;                       // 8*128
  float* seg    = stats + 512 + NBATCH * CCH;        // 8*128

  // init
  zero_f32<<<2048, 256, 0, stream>>>(acc, NPTS * CCH);
  zero_f32<<<8,    256, 0, stream>>>(stats, 2560);
  convert_w<<<1728, 256, 0, stream>>>(W1, Wcol1);
  convert_w<<<1728, 256, 0, stream>>>(W2, Wcol2);

  // conv1 -> BN1 stats -> BN1 apply + relu + segment pool
  conv_wmma<<<KOFF * BLKS_PER_K, 256, 0, stream>>>(x, Wcol1, in_idx, outix,
                                                   nullptr, nullptr, acc);
  bn_stats<<<1024, 256, 0, stream>>>(acc, sums1, sumsq1);
  bn_apply_pool<<<1024, 256, 0, stream>>>(acc, sums1, sumsq1, gamma1, beta1,
                                          bidv, act, pooled);
  // SE
  se_mlp<<<1, 256, 0, stream>>>(pooled, fc1_w, fc2_w, seg);

  // conv2 (SE scale fused into gather) -> BN2 -> residual + relu
  zero_f32<<<2048, 256, 0, stream>>>(acc, NPTS * CCH);
  conv_wmma<<<KOFF * BLKS_PER_K, 256, 0, stream>>>(act, Wcol2, in_idx, outix,
                                                   seg, bidv, acc);
  bn_stats<<<1024, 256, 0, stream>>>(acc, sums2, sumsq2);
  bn_final<<<2048, 256, 0, stream>>>(acc, x, sums2, sumsq2, gamma2, beta2, out);
}
